// PieceGNN_6691559047721
// MI455X (gfx1250) — matrix operand order repaired and checked by reading, not compiled, not run on hardware
//
#include <hip/hip_runtime.h>
#include <math.h>

// ---------------------------------------------------------------------------
// 3-layer GCN for MI455X (gfx1250, wave32).
//   h = X @ W          -> v_wmma_f32_16x16x32_f16 (f32 accumulate, K=256)
//   msg/scatter        -> one wave per edge, 8 ch/lane, global_atomic_add_f32
//                         (feature matrix = 51.2 MB -> resident in 192 MB L2)
//   bias + gelu        -> in-place elementwise
// ---------------------------------------------------------------------------

typedef __attribute__((ext_vector_type(16))) _Float16 v16h;
typedef __attribute__((ext_vector_type(8)))  float    v8f;

#define CH       256
#define KCHUNKS  8          // 256 / 32

// ---------------- degree / normalization ----------------
__global__ void k_init_deg(float* deg, int n) {
  int i = blockIdx.x * blockDim.x + threadIdx.x;
  if (i < n) deg[i] = 1.0f;                      // self-loop contributes 1
}

__global__ void k_count_deg(const int* __restrict__ ei, int E, float* deg) {
  int e = blockIdx.x * blockDim.x + threadIdx.x;
  if (e < E) atomicAdd(&deg[ei[E + e]], 1.0f);   // dst = ei[1][e]
}

__global__ void k_finalize_dinv(float* deg, int n) {
  int i = blockIdx.x * blockDim.x + threadIdx.x;
  if (i < n) deg[i] = rsqrtf(deg[i]);            // deg >= 1 always
}

// ---------------- weight repack: f32 [K=256][N=256] -> f16 B-fragments -----
// Fragment block for (ntile, kc): 32 lanes x 16 halfs, contiguous (1 KB).
// Per lane: n = ntile*16 + (lane&15), half j holds K = kc*32 + (j>>3)*16 + (lane>>4)*8 + (j&7)
// (matches A order: lanes 0-15 K=0..7,16..23, lanes 16-31 K=8..15,24..31, col=lane%16)
__global__ void k_repack_w(const float* __restrict__ W, _Float16* __restrict__ Wf) {
  int i = blockIdx.x * blockDim.x + threadIdx.x;   // 0 .. 65535
  int j     =  i        & 15;
  int lane  = (i >> 4)  & 31;
  int kc    = (i >> 9)  & 7;
  int ntile =  i >> 12;
  int n = ntile * 16 + (lane & 15);
  int k = kc * 32 + ((j >> 3) * 16) + ((lane >> 4) * 8) + (j & 7);
  Wf[i] = (_Float16)W[k * CH + n];
}

// ---------------- WMMA GEMM: H[N,256] = X[N,256] @ W[256,256] --------------
// Block = 256 threads (8 waves) -> 32 rows x 256 cols.
// Wave (rg, cg): rows [r0, r0+16), cols [cg*64, cg*64+64) = 4 WMMA n-tiles.
__global__ void __launch_bounds__(256)
k_gemm_wmma(const float* __restrict__ X, const _Float16* __restrict__ Wf,
            float* __restrict__ H, int nrows) {
  const int lane = threadIdx.x & 31;
  const int wave = threadIdx.x >> 5;
  const int rg   = wave >> 2;                 // 0..1
  const int cg   = wave & 3;                  // 0..3
  const int r0   = blockIdx.x * 32 + rg * 16;
  if (r0 >= nrows) return;                    // wave-uniform (N % 16 == 0)

  const int hi   = lane >> 4;
  const int mrow = r0 + (lane & 15);
  const float* xr = X + (size_t)mrow * CH;

  v8f acc[4] = {v8f{}, v8f{}, v8f{}, v8f{}};

  for (int kc = 0; kc < KCHUNKS; ++kc) {
    // A fragment (16-bit A 16x32 layout):
    //   halfs 0..7  = K kc*32 +      hi*8 + j
    //   halfs 8..15 = K kc*32 + 16 + hi*8 + j
    union { v16h v; _Float16 e[16]; } a;
    const int kb = kc * 32 + hi * 8;
#pragma unroll
    for (int j = 0; j < 8; ++j) a.e[j]     = (_Float16)xr[kb + j];
#pragma unroll
    for (int j = 0; j < 8; ++j) a.e[8 + j] = (_Float16)xr[kb + 16 + j];

#pragma unroll
    for (int t = 0; t < 4; ++t) {
      const int ntile = cg * 4 + t;
      const v16h b = *(const v16h*)(Wf + ((size_t)(ntile * KCHUNKS + kc) * 32 + lane) * 16);
      acc[t] = __builtin_amdgcn_wmma_f32_16x16x32_f16(
          false, a.v, false, b, (short)0, acc[t], false, false);
    }
  }

  // C/D layout: vgpr r -> row r0 + hi*8 + r, col = ntile*16 + (lane&15)
#pragma unroll
  for (int t = 0; t < 4; ++t) {
    const int col = (cg * 4 + t) * 16 + (lane & 15);
    union { v8f v; float e[8]; } c; c.v = acc[t];
#pragma unroll
    for (int r = 0; r < 8; ++r) {
      const int row = r0 + hi * 8 + r;
      H[(size_t)row * CH + col] = c.e[r];
    }
  }
}

// ---------------- aggregation ----------------
// T[n][c] = dinv[n]^2 * H[n][c]   (self-loop term; full overwrite -> no memset)
__global__ void k_self_init(const float* __restrict__ H, const float* __restrict__ dinv,
                            float* __restrict__ T, int total) {
  int idx = blockIdx.x * blockDim.x + threadIdx.x;
  if (idx < total) {
    float d = dinv[idx >> 8];
    T[idx] = d * d * H[idx];
  }
}

// One wave per edge; lane handles 8 channels (coalesced 1 KB row read,
// 8x global_atomic_add_f32 into the L2-resident accumulator).
__global__ void __launch_bounds__(256)
k_edge_agg(const int* __restrict__ ei, int E, const float* __restrict__ dinv,
           const float* __restrict__ H, float* __restrict__ T) {
  const int e    = (int)((blockIdx.x * (size_t)blockDim.x + threadIdx.x) >> 5);
  const int lane = threadIdx.x & 31;
  if (e >= E) return;
  const int src = ei[e];
  const int dst = ei[E + e];
  const float nrm = dinv[src] * dinv[dst];

  const float4* hs = (const float4*)(H + (size_t)src * CH) + lane * 2;
  const float4 h0 = hs[0];
  const float4 h1 = hs[1];
  float* td = T + (size_t)dst * CH + lane * 8;
  atomicAdd(td + 0, nrm * h0.x);
  atomicAdd(td + 1, nrm * h0.y);
  atomicAdd(td + 2, nrm * h0.z);
  atomicAdd(td + 3, nrm * h0.w);
  atomicAdd(td + 4, nrm * h1.x);
  atomicAdd(td + 5, nrm * h1.y);
  atomicAdd(td + 6, nrm * h1.z);
  atomicAdd(td + 7, nrm * h1.w);
}

// ---------------- bias + (tanh-approx) GELU, in place ----------------
__global__ void k_bias_act(float* __restrict__ O, const float* __restrict__ b,
                           int total, int applyGelu) {
  int idx = blockIdx.x * blockDim.x + threadIdx.x;
  if (idx < total) {
    float v = O[idx] + b[idx & (CH - 1)];
    if (applyGelu) {
      float v3 = v * v * v;
      v = 0.5f * v * (1.0f + tanhf(0.7978845608f * (v + 0.044715f * v3)));
    }
    O[idx] = v;
  }
}

// ---------------------------------------------------------------------------
extern "C" void kernel_launch(void* const* d_in, const int* in_sizes, int n_in,
                              void* d_out, int out_size, void* d_ws, size_t ws_size,
                              hipStream_t stream) {
  (void)n_in; (void)out_size; (void)ws_size;
  const float* x  = (const float*)d_in[0];
  const int*   ei = (const int*)  d_in[1];   // [2,E]: src row 0, dst row 1
  // d_in[2] = batch (unused)
  const float* W[3] = { (const float*)d_in[3], (const float*)d_in[5], (const float*)d_in[7] };
  const float* b[3] = { (const float*)d_in[4], (const float*)d_in[6], (const float*)d_in[8] };

  const int N = in_sizes[0] / CH;
  const int E = in_sizes[1] / 2;
  const int total = N * CH;

  // workspace carve-out (256B aligned): dinv[N] + 3x Wf(128KB) + H[N*256] ~ 52.5 MB
  char* ws = (char*)d_ws;
  size_t off = 0;
  auto carve = [&](size_t bytes) -> char* {
    char* p = ws + off;
    off = (off + bytes + 255) & ~(size_t)255;
    return p;
  };
  float*    dinv = (float*)carve((size_t)N * sizeof(float));
  _Float16* Wf[3];
  for (int i = 0; i < 3; ++i) Wf[i] = (_Float16*)carve((size_t)CH * CH * sizeof(_Float16));
  float*    H = (float*)carve((size_t)total * sizeof(float));
  float*    T = (float*)d_out;               // accumulator / activations live in d_out

  const int B256 = 256;

  // degree -> dinv
  k_init_deg     <<<(N + B256 - 1) / B256, B256, 0, stream>>>(dinv, N);
  k_count_deg    <<<(E + B256 - 1) / B256, B256, 0, stream>>>(ei, E, dinv);
  k_finalize_dinv<<<(N + B256 - 1) / B256, B256, 0, stream>>>(dinv, N);

  // repack weights to f16 WMMA B-fragments (done every call: deterministic)
  for (int i = 0; i < 3; ++i)
    k_repack_w<<<(CH * CH) / B256, B256, 0, stream>>>(W[i], Wf[i]);

  const int gemmBlocks = (N + 31) / 32;
  const int edgeBlocks = (int)(((size_t)E * 32 + B256 - 1) / B256);
  const int elemBlocks = (total + B256 - 1) / B256;

  for (int layer = 0; layer < 3; ++layer) {
    const float* feat = (layer == 0) ? x : (const float*)T;
    k_gemm_wmma<<<gemmBlocks, B256, 0, stream>>>(feat, Wf[layer], H, N);
    k_self_init<<<elemBlocks, B256, 0, stream>>>(H, dinv, T, total);
    k_edge_agg <<<edgeBlocks, B256, 0, stream>>>(ei, E, dinv, H, T);
    k_bias_act <<<elemBlocks, B256, 0, stream>>>(T, b[layer], total, layer < 2 ? 1 : 0);
  }
}
